// UDNO_77197742178606
// MI455X (gfx1250) — matrix operand order repaired and checked by reading, not compiled, not run on hardware
//
#include <hip/hip_runtime.h>
#include <math.h>

typedef __bf16 bf16;
typedef __attribute__((ext_vector_type(16))) __bf16 v16bf;
typedef __attribute__((ext_vector_type(8)))  float  v8f;

#ifndef M_PI
#define M_PI 3.14159265358979323846
#endif

// All bf16 activation tensors use a zero halo of 3 pixels:
//   layout [n][H+6][W+6][C], pixel (y,x) at ((y+3)*(W+6)+(x+3))*C.
// This makes the WMMA conv inner loop fully branchless.

// ---------------------------------------------------------------------------
// DISCO psi basis (9 x 7 x 7), computed once on device (tiny, 1 thread).
// ---------------------------------------------------------------------------
__global__ void psi_kernel(float* __restrict__ psi) {
  if (threadIdx.x != 0 || blockIdx.x != 0) return;
  const double rpix = 3.2;
  const double dr   = rpix / 2.0;          // nr - 1 = 2
  const double dphi = 2.0 * M_PI / 4.0;    // nphi = 4
  double bas[9][49];
  for (int iy = 0; iy < 7; ++iy) {
    for (int ix = 0; ix < 7; ++ix) {
      const int cell = iy * 7 + ix;
      const double dy = (double)(iy - 3), dx = (double)(ix - 3);
      const double r = sqrt(dy * dy + dx * dx);
      double a = atan2(dy, dx);
      double phi = (a < 0.0) ? a + 2.0 * M_PI : a;   // np.mod(...,2pi)
      const double mask = (r <= rpix) ? 1.0 : 0.0;
      double c0 = 1.0 - r / dr; if (c0 < 0.0) c0 = 0.0;
      bas[0][cell] = c0 * mask;
      for (int i = 1; i <= 2; ++i) {
        double rad = 1.0 - fabs(r - i * dr) / dr; if (rad < 0.0) rad = 0.0;
        rad *= mask;
        for (int j = 0; j < 4; ++j) {
          double m = fmod(phi - j * dphi + M_PI, 2.0 * M_PI);
          if (m < 0.0) m += 2.0 * M_PI;              // python mod semantics
          double d = fabs(m - M_PI);
          double ang = 1.0 - d / dphi; if (ang < 0.0) ang = 0.0;
          bas[1 + (i - 1) * 4 + j][cell] = rad * ang;
        }
      }
    }
  }
  for (int b = 0; b < 9; ++b) {
    double s = 0.0;
    for (int c = 0; c < 49; ++c) s += bas[b][c];
    const double inv = (s > 0.0) ? 1.0 / s : 1.0;
    for (int c = 0; c < 49; ++c) psi[b * 49 + c] = (float)(bas[b][c] * inv);
  }
}

// ---------------------------------------------------------------------------
// Expand w(O,I,9) x psi(9,49) -> bf16 filter bank [49][O][Cp], zero padded.
// ---------------------------------------------------------------------------
__global__ void wt_kernel(const float* __restrict__ w, const float* __restrict__ psi,
                          bf16* __restrict__ f, int O, int I, int Cp) {
  long idx = (long)blockIdx.x * blockDim.x + threadIdx.x;
  long total = (long)49 * O * Cp;
  if (idx >= total) return;
  int c  = (int)(idx % Cp);
  int o  = (int)((idx / Cp) % O);
  int kk = (int)(idx / ((long)Cp * O));
  float acc = 0.f;
  if (c < I) {
    const float* wr = w + ((size_t)o * I + c) * 9;
#pragma unroll
    for (int k = 0; k < 9; ++k) acc += wr[k] * psi[k * 49 + kk];
  }
  f[idx] = (bf16)acc;
}

// image (2,2,320,320) NCHW f32 -> haloed NHWC bf16 [2][326][326][32]
__global__ void img_kernel(const float* __restrict__ img, bf16* __restrict__ dst) {
  size_t idx = (size_t)blockIdx.x * blockDim.x + threadIdx.x;
  size_t total = (size_t)2 * 326 * 326 * 32;
  if (idx >= total) return;
  int c  = (int)(idx % 32);
  size_t t = idx / 32;
  int xx = (int)(t % 326); t /= 326;
  int yy = (int)(t % 326);
  int n  = (int)(t / 326);
  float v = 0.f;
  if (yy >= 3 && yy < 323 && xx >= 3 && xx < 323 && c < 2)
    v = img[((size_t)n * 2 + c) * 102400 + (size_t)(yy - 3) * 320 + (xx - 3)];
  dst[idx] = (bf16)v;
}

// ---------------------------------------------------------------------------
// Implicit-GEMM 7x7 conv, V_WMMA_F32_16X16X32_BF16, 2x2 register blocking.
//   in  : bf16 haloed NHWC [2][H+6][W+6][Cp]
//   wgt : bf16 [49][O][Cp]
//   out : f32 linear NHWC [2][H][W][O]
// Block = 128 threads (4 waves). Each wave: 32 out-ch x 32 px (4 WMMAs/iter).
// Block tile = 32 out-ch x 128 px. Zero halo -> branchless inner loop; the
// per-tap shift is one uniform offset (dy*Wp+dx)*Cp added to a fixed base.
// ---------------------------------------------------------------------------
__global__ __launch_bounds__(128) void conv7_wmma(
    const bf16* __restrict__ in, const bf16* __restrict__ wgt,
    float* __restrict__ out, int H, int W, int Cp, int O) {
  const int HW = H * W;
  const int Wp = W + 6;
  const int lane = threadIdx.x & 31;
  const int wave = threadIdx.x >> 5;
  const int lh = lane & 15;
  const int hi = lane >> 4;                 // 0 | 1
  const int n  = blockIdx.z;
  const int oB = blockIdx.y * 32;
  const int pw = blockIdx.x * 128 + wave * 32;
  const int p0 = pw + lh;
  const int p1 = p0 + 16;
  const int q0 = (p0 < HW) ? p0 : HW - 1;   // clamp: halo keeps loads safe
  const int q1 = (p1 < HW) ? p1 : HW - 1;
  const int y0 = q0 / W, x0 = q0 - y0 * W;
  const int y1 = q1 / W, x1 = q1 - y1 * W;
  const bf16* inN = in + (size_t)n * (H + 6) * Wp * Cp;
  const bf16* b0base = inN + ((size_t)(y0 + 3) * Wp + (x0 + 3)) * Cp;
  const bf16* b1base = inN + ((size_t)(y1 + 3) * Wp + (x1 + 3)) * Cp;
  const bf16* a0row = wgt + (size_t)(oB + lh) * Cp;       // rows oB..oB+15
  const bf16* a1row = a0row + (size_t)16 * Cp;            // rows oB+16..oB+31
  const size_t astep = (size_t)O * Cp;

  v8f acc00 = {}, acc01 = {}, acc10 = {}, acc11 = {};
  for (int kk = 0; kk < 49; ++kk) {
    const int q  = kk / 7;
    const long koff = (long)((q - 3) * Wp + (kk - q * 7) - 3) * Cp;
    const bf16* b0s = b0base + koff;
    const bf16* b1s = b1base + koff;
    const bf16* a0s = a0row + (size_t)kk * astep;
    const bf16* a1s = a1row + (size_t)kk * astep;
    if (kk + 1 < 49) __builtin_prefetch(a0s + astep, 0, 1);  // global_prefetch_b8
    for (int cc = 0; cc < Cp; cc += 32) {
      v16bf a0, a1, b0, b1;
      {
        uint4* v = reinterpret_cast<uint4*>(&a0);
        v[0] = *reinterpret_cast<const uint4*>(a0s + cc + (hi << 3));
        v[1] = *reinterpret_cast<const uint4*>(a0s + cc + 16 + (hi << 3));
        v = reinterpret_cast<uint4*>(&a1);
        v[0] = *reinterpret_cast<const uint4*>(a1s + cc + (hi << 3));
        v[1] = *reinterpret_cast<const uint4*>(a1s + cc + 16 + (hi << 3));
        v = reinterpret_cast<uint4*>(&b0);
        v[0] = *reinterpret_cast<const uint4*>(b0s + cc + (hi << 3));
        v[1] = *reinterpret_cast<const uint4*>(b0s + cc + 16 + (hi << 3));
        v = reinterpret_cast<uint4*>(&b1);
        v[0] = *reinterpret_cast<const uint4*>(b1s + cc + (hi << 3));
        v[1] = *reinterpret_cast<const uint4*>(b1s + cc + 16 + (hi << 3));
      }
      acc00 = __builtin_amdgcn_wmma_f32_16x16x32_bf16(false, a0, false, b0,
                                                      (short)0, acc00, false, false);
      acc01 = __builtin_amdgcn_wmma_f32_16x16x32_bf16(false, a0, false, b1,
                                                      (short)0, acc01, false, false);
      acc10 = __builtin_amdgcn_wmma_f32_16x16x32_bf16(false, a1, false, b0,
                                                      (short)0, acc10, false, false);
      acc11 = __builtin_amdgcn_wmma_f32_16x16x32_bf16(false, a1, false, b1,
                                                      (short)0, acc11, false, false);
    }
  }
  // D layout: lane<16 -> ch r, lane>=16 -> ch 8+r; N = lane%16 (= our pixel).
  {
    float* base = out + (size_t)n * HW * O + oB + (hi << 3);
    if (p0 < HW) {
      float* op = base + (size_t)p0 * O;
      *reinterpret_cast<float4*>(op)      = make_float4(acc00[0], acc00[1], acc00[2], acc00[3]);
      *reinterpret_cast<float4*>(op + 4)  = make_float4(acc00[4], acc00[5], acc00[6], acc00[7]);
      *reinterpret_cast<float4*>(op + 16) = make_float4(acc10[0], acc10[1], acc10[2], acc10[3]);
      *reinterpret_cast<float4*>(op + 20) = make_float4(acc10[4], acc10[5], acc10[6], acc10[7]);
    }
    if (p1 < HW) {
      float* op = base + (size_t)p1 * O;
      *reinterpret_cast<float4*>(op)      = make_float4(acc01[0], acc01[1], acc01[2], acc01[3]);
      *reinterpret_cast<float4*>(op + 4)  = make_float4(acc01[4], acc01[5], acc01[6], acc01[7]);
      *reinterpret_cast<float4*>(op + 16) = make_float4(acc11[0], acc11[1], acc11[2], acc11[3]);
      *reinterpret_cast<float4*>(op + 20) = make_float4(acc11[4], acc11[5], acc11[6], acc11[7]);
    }
  }
}

// ---------------------------------------------------------------------------
// Instance norm pass 1: per (n,c) sum/sumsq via LDS atomics + global atomics.
// x is the linear f32 conv output [n][HW][C].
// ---------------------------------------------------------------------------
__global__ void in_reduce(const float* __restrict__ x, float* __restrict__ stats,
                          int HW, int C, int elemsPerBlock) {
  __shared__ float ssum[512];
  __shared__ float ssq[512];
  for (int c = threadIdx.x; c < C; c += blockDim.x) { ssum[c] = 0.f; ssq[c] = 0.f; }
  __syncthreads();
  const int n = blockIdx.y;
  const size_t total = (size_t)HW * C;
  const size_t start = (size_t)blockIdx.x * elemsPerBlock;
  size_t end = start + elemsPerBlock; if (end > total) end = total;
  const float* xn = x + (size_t)n * total;
  for (size_t e = start + threadIdx.x; e < end; e += blockDim.x) {
    float v = xn[e];
    int c = (int)(e % (size_t)C);
    atomicAdd(&ssum[c], v);
    atomicAdd(&ssq[c], v * v);
  }
  __syncthreads();
  for (int c = threadIdx.x; c < C; c += blockDim.x) {
    atomicAdd(&stats[((size_t)n * 512 + c) * 2 + 0], ssum[c]);
    atomicAdd(&stats[((size_t)n * 512 + c) * 2 + 1], ssq[c]);
  }
}

// pass 2: normalize + LeakyReLU(0.2); writes haloed bf16 (zeros in halo)
__global__ void in_apply(const float* __restrict__ x, const float* __restrict__ stats,
                         bf16* __restrict__ y, int H, int W, int C) {
  const int Hp = H + 6, Wp = W + 6;
  size_t idx = (size_t)blockIdx.x * blockDim.x + threadIdx.x;
  size_t total = (size_t)2 * Hp * Wp * C;
  if (idx >= total) return;
  int c = (int)(idx % C);
  size_t t = idx / C;
  int xx = (int)(t % Wp); t /= Wp;
  int yy = (int)(t % Hp);
  int n  = (int)(t / Hp);
  float r = 0.f;
  if (yy >= 3 && yy < H + 3 && xx >= 3 && xx < W + 3) {
    float invHW = 1.f / (float)(H * W);
    float mean = stats[((size_t)n * 512 + c) * 2 + 0] * invHW;
    float msq  = stats[((size_t)n * 512 + c) * 2 + 1] * invHW;
    float var  = msq - mean * mean;
    size_t src = ((size_t)n * H * W + (size_t)(yy - 3) * W + (xx - 3)) * C + c;
    float v = (x[src] - mean) * rsqrtf(var + 1e-5f);
    r = (v >= 0.f) ? v : 0.2f * v;
  }
  y[idx] = (bf16)r;
}

// 2x2 average pool, haloed bf16 -> haloed bf16
__global__ void pool2(const bf16* __restrict__ x, bf16* __restrict__ y,
                      int H, int W, int C) {
  const int H2 = H >> 1, W2 = W >> 1;
  const int Hp = H + 6, Wp = W + 6, Hp2 = H2 + 6, Wp2 = W2 + 6;
  size_t idx = (size_t)blockIdx.x * blockDim.x + threadIdx.x;
  size_t total = (size_t)2 * Hp2 * Wp2 * C;
  if (idx >= total) return;
  int c = (int)(idx % C);
  size_t t = idx / C;
  int xx = (int)(t % Wp2); t /= Wp2;
  int yy = (int)(t % Hp2);
  int n  = (int)(t / Hp2);
  float r = 0.f;
  if (yy >= 3 && yy < H2 + 3 && xx >= 3 && xx < W2 + 3) {
    const bf16* b = x + (((size_t)n * Hp + 2 * (yy - 3) + 3) * Wp + 2 * (xx - 3) + 3) * C + c;
    r = 0.25f * ((float)b[0] + (float)b[C] +
                 (float)b[(size_t)Wp * C] + (float)b[(size_t)Wp * C + C]);
  }
  y[idx] = (bf16)r;
}

// 2x bilinear upsample, align_corners, haloed bf16 -> haloed bf16
__global__ void up2(const bf16* __restrict__ x, bf16* __restrict__ y,
                    int H, int W, int C) {
  const int H2 = H * 2, W2 = W * 2;
  const int Hp = H + 6, Wp = W + 6, Hp2 = H2 + 6, Wp2 = W2 + 6;
  size_t idx = (size_t)blockIdx.x * blockDim.x + threadIdx.x;
  size_t total = (size_t)2 * Hp2 * Wp2 * C;
  if (idx >= total) return;
  int c = (int)(idx % C);
  size_t t = idx / C;
  int xx = (int)(t % Wp2); t /= Wp2;
  int yy = (int)(t % Hp2);
  int n  = (int)(t / Hp2);
  float r = 0.f;
  if (yy >= 3 && yy < H2 + 3 && xx >= 3 && xx < W2 + 3) {
    int Y = yy - 3, X = xx - 3;
    float py = (float)Y * (float)(H - 1) / (float)(H2 - 1);
    float px = (float)X * (float)(W - 1) / (float)(W2 - 1);
    int y0 = (int)floorf(py), x0 = (int)floorf(px);
    int y1 = (y0 + 1 < H) ? y0 + 1 : H - 1;
    int x1 = (x0 + 1 < W) ? x0 + 1 : W - 1;
    float ty = py - (float)y0, tx = px - (float)x0;
    const bf16* bp = x + (size_t)n * Hp * Wp * C + c;
    float v00 = (float)bp[((size_t)(y0 + 3) * Wp + x0 + 3) * C];
    float v01 = (float)bp[((size_t)(y0 + 3) * Wp + x1 + 3) * C];
    float v10 = (float)bp[((size_t)(y1 + 3) * Wp + x0 + 3) * C];
    float v11 = (float)bp[((size_t)(y1 + 3) * Wp + x1 + 3) * C];
    float v0 = v00 + (v01 - v00) * tx;
    float v1 = v10 + (v11 - v10) * tx;
    r = v0 + (v1 - v0) * ty;
  }
  y[idx] = (bf16)r;
}

// channel concat of two haloed tensors (halos are zero in both sources)
__global__ void catk(const bf16* __restrict__ a, const bf16* __restrict__ b,
                     bf16* __restrict__ d, int H, int W, int C1, int C2) {
  const int Hp = H + 6, Wp = W + 6;
  const int C = C1 + C2;
  size_t idx = (size_t)blockIdx.x * blockDim.x + threadIdx.x;
  size_t total = (size_t)2 * Hp * Wp * C;
  if (idx >= total) return;
  int c = (int)(idx % C);
  size_t t = idx / C;
  size_t pix = t % ((size_t)Hp * Wp);
  int n = (int)(t / ((size_t)Hp * Wp));
  d[idx] = (c < C1) ? a[((size_t)n * Hp * Wp + pix) * C1 + c]
                    : b[((size_t)n * Hp * Wp + pix) * C2 + (c - C1)];
}

// final 1x1 conv (32 -> 2) + bias; haloed bf16 in, NCHW f32 out
__global__ void finalk(const bf16* __restrict__ act, const float* __restrict__ fw,
                       const float* __restrict__ fb, float* __restrict__ out, int HW) {
  int idx = blockIdx.x * blockDim.x + threadIdx.x;
  int total = 2 * 2 * HW;
  if (idx >= total) return;
  int p = idx % HW;
  int o = (idx / HW) % 2;
  int n = idx / (2 * HW);
  int y = p / 320, x = p - y * 320;
  const bf16* a = act + (((size_t)n * 326 + y + 3) * 326 + x + 3) * 32;
  float s = fb[o];
#pragma unroll
  for (int i = 0; i < 32; ++i) s += fw[o * 32 + i] * (float)a[i];
  out[idx] = s;
}

// ---------------------------------------------------------------------------
// Host orchestration (~160 MB carved from d_ws).
// Input ordering: jax tree_leaves (alphabetical) assumed, deterministic
// fallback to insertion order via in_sizes[1].
// ---------------------------------------------------------------------------
extern "C" void kernel_launch(void* const* d_in, const int* in_sizes, int n_in,
                              void* d_out, int out_size, void* d_ws, size_t ws_size,
                              hipStream_t stream) {
  (void)out_size; (void)ws_size;
  int iIMG = 0, iBNA, iBNB, iD[4][2], iFB, iFW, iUP[4][2], iUT[4];
  if (n_in >= 2 && in_sizes[1] == 32 * 2 * 9) {
    iD[0][0]=1; iD[0][1]=2; iD[1][0]=3; iD[1][1]=4;
    iD[2][0]=5; iD[2][1]=6; iD[3][0]=7; iD[3][1]=8;
    iBNA=9; iBNB=10;
    iUT[0]=11; iUT[1]=12; iUT[2]=13; iUT[3]=14;
    iUP[0][0]=15; iUP[0][1]=16; iUP[1][0]=17; iUP[1][1]=18;
    iUP[2][0]=19; iUP[2][1]=20; iUP[3][0]=21; iUP[3][1]=22;
    iFW=23; iFB=24;
  } else {
    iBNA=1; iBNB=2;
    iD[0][0]=3; iD[0][1]=4; iD[1][0]=5; iD[1][1]=6;
    iD[2][0]=7; iD[2][1]=8; iD[3][0]=9; iD[3][1]=10;
    iFB=11; iFW=12;
    iUP[0][0]=13; iUP[0][1]=14; iUP[1][0]=15; iUP[1][1]=16;
    iUP[2][0]=17; iUP[2][1]=18; iUP[3][0]=19; iUP[3][1]=20;
    iUT[0]=21; iUT[1]=22; iUT[2]=23; iUT[3]=24;
  }

  size_t off = 0;
  auto carve = [&](size_t bytes) -> char* {
    char* p = (char*)d_ws + off;
    off += (bytes + 255) & ~(size_t)255;
    return p;
  };
  float* psi   = (float*)carve(441 * 4);
  float* stats = (float*)carve((size_t)2 * 512 * 2 * 4);
  bf16*  wbuf  = (bf16*) carve((size_t)49 * 512 * 512 * 2);       // 25.7 MB
  bf16*  bufA  = (bf16*) carve((size_t)2 * 326 * 326 * 64 * 2);   // 27.2 MB
  bf16*  bufB  = (bf16*) carve((size_t)2 * 326 * 326 * 64 * 2);
  bf16*  bufC  = (bf16*) carve((size_t)2 * 326 * 326 * 64 * 2);
  float* convf = (float*)carve((size_t)2 * 320 * 320 * 32 * 4);   // 26.2 MB
  bf16*  sk0   = (bf16*) carve((size_t)2 * 326 * 326 * 32 * 2);
  bf16*  sk1   = (bf16*) carve((size_t)2 * 166 * 166 * 64 * 2);
  bf16*  sk2   = (bf16*) carve((size_t)2 * 86 * 86 * 128 * 2);
  bf16*  sk3   = (bf16*) carve((size_t)2 * 46 * 46 * 256 * 2);

  auto elw = [](size_t tot) { return (unsigned)((tot + 255) / 256); };
  auto htot = [](int H, int W, int C) { return (size_t)2 * (H + 6) * (W + 6) * C; };

  psi_kernel<<<1, 32, 0, stream>>>(psi);
  img_kernel<<<elw(htot(320, 320, 32)), 256, 0, stream>>>((const float*)d_in[iIMG], bufA);

  // DISCO conv -> instance norm -> leaky for one conv layer
  auto convnorm = [&](int wi, int O, int I, int H, int W, const bf16* src, bf16* dst) {
    int Cp = (I < 32) ? 32 : I;
    long tw = (long)49 * O * Cp;
    wt_kernel<<<(unsigned)((tw + 255) / 256), 256, 0, stream>>>(
        (const float*)d_in[wi], psi, wbuf, O, I, Cp);
    dim3 g((H * W + 127) / 128, O / 32, 2);
    conv7_wmma<<<g, 128, 0, stream>>>(src, wbuf, convf, H, W, Cp, O);
    (void)hipMemsetAsync(stats, 0, (size_t)2 * 512 * 2 * sizeof(float), stream);
    size_t tot = (size_t)H * W * O;
    const int elems = 256 * 32;
    dim3 gr((unsigned)((tot + elems - 1) / elems), 2);
    in_reduce<<<gr, 256, 0, stream>>>(convf, stats, H * W, O, elems);
    in_apply<<<elw(htot(H, W, O)), 256, 0, stream>>>(convf, stats, dst, H, W, O);
  };

  // ---------------- encoder ----------------
  convnorm(iD[0][0], 32, 2, 320, 320, bufA, bufB);
  convnorm(iD[0][1], 32, 32, 320, 320, bufB, sk0);
  pool2<<<elw(htot(160, 160, 32)), 256, 0, stream>>>(sk0, bufA, 320, 320, 32);

  convnorm(iD[1][0], 64, 32, 160, 160, bufA, bufB);
  convnorm(iD[1][1], 64, 64, 160, 160, bufB, sk1);
  pool2<<<elw(htot(80, 80, 64)), 256, 0, stream>>>(sk1, bufA, 160, 160, 64);

  convnorm(iD[2][0], 128, 64, 80, 80, bufA, bufB);
  convnorm(iD[2][1], 128, 128, 80, 80, bufB, sk2);
  pool2<<<elw(htot(40, 40, 128)), 256, 0, stream>>>(sk2, bufA, 80, 80, 128);

  convnorm(iD[3][0], 256, 128, 40, 40, bufA, bufB);
  convnorm(iD[3][1], 256, 256, 40, 40, bufB, sk3);
  pool2<<<elw(htot(20, 20, 256)), 256, 0, stream>>>(sk3, bufA, 40, 40, 256);

  // ---------------- bottleneck ----------------
  convnorm(iBNA, 512, 256, 20, 20, bufA, bufB);
  convnorm(iBNB, 512, 512, 20, 20, bufB, bufA);   // bufA: 512 @ 20

  // ---------------- decoder ----------------
  // level 3 (deepest)
  up2<<<elw(htot(40, 40, 512)), 256, 0, stream>>>(bufA, bufB, 20, 20, 512);
  convnorm(iUT[0], 256, 512, 40, 40, bufB, bufA);
  catk<<<elw(htot(40, 40, 512)), 256, 0, stream>>>(bufA, sk3, bufC, 40, 40, 256, 256);
  convnorm(iUP[0][0], 256, 512, 40, 40, bufC, bufA);
  convnorm(iUP[0][1], 256, 256, 40, 40, bufA, bufB);  // bufB: 256 @ 40

  // level 2
  up2<<<elw(htot(80, 80, 256)), 256, 0, stream>>>(bufB, bufA, 40, 40, 256);
  convnorm(iUT[1], 128, 256, 80, 80, bufA, bufB);
  catk<<<elw(htot(80, 80, 256)), 256, 0, stream>>>(bufB, sk2, bufC, 80, 80, 128, 128);
  convnorm(iUP[1][0], 128, 256, 80, 80, bufC, bufA);
  convnorm(iUP[1][1], 128, 128, 80, 80, bufA, bufB);  // bufB: 128 @ 80

  // level 1
  up2<<<elw(htot(160, 160, 128)), 256, 0, stream>>>(bufB, bufA, 80, 80, 128);
  convnorm(iUT[2], 64, 128, 160, 160, bufA, bufB);
  catk<<<elw(htot(160, 160, 128)), 256, 0, stream>>>(bufB, sk1, bufC, 160, 160, 64, 64);
  convnorm(iUP[2][0], 64, 128, 160, 160, bufC, bufA);
  convnorm(iUP[2][1], 64, 64, 160, 160, bufA, bufB);  // bufB: 64 @ 160

  // level 0
  up2<<<elw(htot(320, 320, 64)), 256, 0, stream>>>(bufB, bufA, 160, 160, 64);
  convnorm(iUT[3], 32, 64, 320, 320, bufA, bufB);
  catk<<<elw(htot(320, 320, 64)), 256, 0, stream>>>(bufB, sk0, bufC, 320, 320, 32, 32);
  convnorm(iUP[3][0], 32, 64, 320, 320, bufC, bufA);
  convnorm(iUP[3][1], 32, 32, 320, 320, bufA, bufB);  // bufB: 32 @ 320

  // final 1x1 conv -> NCHW f32
  finalk<<<elw((size_t)2 * 2 * 102400), 256, 0, stream>>>(
      bufB, (const float*)d_in[iFW], (const float*)d_in[iFB], (float*)d_out, 102400);
}
